// CharToWord_10325101379850
// MI455X (gfx1250) — compile-verified, matrix operand order checked by reading, not run on hardware
//
#include <hip/hip_runtime.h>
#include <math.h>

#define T_MAX 20
#define EMBD  64
#define HID   128
#define B_TOT 32768
#define SPITCH 264   // staged attn row pitch in halves (512B row + 16B TDM pad)

typedef _Float16 v8h  __attribute__((ext_vector_type(8)));
typedef _Float16 v16h __attribute__((ext_vector_type(16)));
typedef float    v8f  __attribute__((ext_vector_type(8)));
typedef unsigned int uint32x4 __attribute__((ext_vector_type(4)));
typedef int          int32x4  __attribute__((ext_vector_type(4)));
typedef int          int32x8  __attribute__((ext_vector_type(8)));

#if defined(__has_builtin)
#if __has_builtin(__builtin_amdgcn_tensor_load_to_lds)
#define HAVE_TDM 1
#endif
#endif
#ifndef HAVE_TDM
#define HAVE_TDM 0
#endif

__device__ inline v8f splat8(float x){
    v8f r;
#pragma unroll
    for (int i = 0; i < 8; ++i) r[i] = x;
    return r;
}

// Load one 16x32 (A) or 32x16 (B, from row-major W so B = W^T) f16 WMMA fragment.
// CDNA5 layout: lane halves 0..7 -> K = kbase + hi*8 + 0..7,
//               halves 8..15     -> K = kbase + 16 + hi*8 + 0..7.
__device__ inline v16h load_frag(const _Float16* base, int row, int ld, int kbase, int hi){
    const _Float16* p = base + (size_t)row * ld + kbase + hi * 8;
    v8h lo = *(const v8h*)p;
    v8h hv = *(const v8h*)(p + 16);
    v16h r;
#pragma unroll
    for (int i = 0; i < 8; ++i) { r[i] = lo[i]; r[i + 8] = hv[i]; }
    return r;
}

__device__ inline v8f wmma16(v16h a, v16h b, v8f c){
    return __builtin_amdgcn_wmma_f32_16x16x32_f16(false, a, false, b, (short)0, c, false, false);
}

__device__ inline float sigm(float x){ return 1.f / (1.f + __expf(-x)); }

// ---------------------------------------------------------------------------
// GRU kernel: one direction. 8 waves/WG, 32 words per wave (two 16-row tiles
// sharing every Whh B-fragment), 256 words per WG.
// Input projection is a precomputed [128 chars][384] f16 table (embP) built
// once per WG with WMMA, so the recurrence does only h @ Whh^T on WMMA.
// ws layout: [chunk_words][T_MAX][256] f16; fwd -> cols 0..127, bwd -> 128..255
// ---------------------------------------------------------------------------
__global__ __launch_bounds__(256)
void gru_dir_kernel(const int* __restrict__ chars, const int* __restrict__ lens,
                    const float* __restrict__ emb, const float* __restrict__ Wih,
                    const float* __restrict__ Whh, const float* __restrict__ bih,
                    const float* __restrict__ bhh, _Float16* __restrict__ ws,
                    int word_base, int dir)
{
    extern __shared__ char smem[];
    _Float16* s_embP  = (_Float16*)(smem);            // 128*384 f16 = 98304 B
    _Float16* s_Whh16 = (_Float16*)(smem + 98304);    // 384*128 f16 = 98304 B
    char*     s_tmp   = smem + 196608;                // 65536 B: staging then hbuf
    _Float16* s_emb16 = (_Float16*)(s_tmp);           // 128*64 f16 (staging)
    _Float16* s_Wih16 = (_Float16*)(s_tmp + 16384);   // 384*64 f16 (staging)
    float*    s_bhh   = (float*)(smem + 262144);      // 384 f32
    int*      s_ch    = (int*)  (smem + 263680);      // 256*20 ints
    int*      s_len   = (int*)  (smem + 284160);      // 256 ints

    const int tid  = threadIdx.x;
    const int wgw0 = blockIdx.x * 256;   // local (chunk-relative) word base of this WG

    for (int i = tid; i < 128 * EMBD; i += 256) s_emb16[i] = (_Float16)emb[i];
    for (int i = tid; i < 384 * EMBD; i += 256) s_Wih16[i] = (_Float16)Wih[i];
    for (int i = tid; i < 384 * HID;  i += 256) s_Whh16[i] = (_Float16)Whh[i];
    for (int i = tid; i < 384;        i += 256) s_bhh[i]   = bhh[i];
    for (int i = tid; i < 256 * T_MAX; i += 256) {
        int wl = i / T_MAX;
        s_ch[i] = chars[(size_t)(word_base + wgw0 + wl) * T_MAX + (i % T_MAX)];
    }
    if (tid < 256) s_len[tid] = lens[word_base + wgw0 + tid];
    __syncthreads();

    const int wave = tid >> 5;
    const int lane = tid & 31;
    const int nl   = lane & 15;
    const int hi   = lane >> 4;

    // ---- build embP = emb @ Wih^T + bih for this wave's 16 chars ----
    {
        v16h a0 = load_frag(s_emb16, wave * 16 + nl, EMBD, 0,  hi);
        v16h a1 = load_frag(s_emb16, wave * 16 + nl, EMBD, 32, hi);
#pragma unroll 1
        for (int nt = 0; nt < 24; ++nt) {
            v8f acc = splat8(bih[nt * 16 + nl]);
            acc = wmma16(a0, load_frag(s_Wih16, nt * 16 + nl, EMBD, 0,  hi), acc);
            acc = wmma16(a1, load_frag(s_Wih16, nt * 16 + nl, EMBD, 32, hi), acc);
#pragma unroll
            for (int v = 0; v < 8; ++v)
                s_embP[(wave * 16 + v + 8 * hi) * 384 + nt * 16 + nl] = (_Float16)acc[v];
        }
    }
    __syncthreads();   // everyone done reading staging before it becomes hbuf

    _Float16* hbuf = (_Float16*)(s_tmp + wave * 8192);   // [32][128] f16 per wave
    {
        v8h z;
#pragma unroll
        for (int i = 0; i < 8; ++i) z[i] = (_Float16)0.f;
        for (int i = lane * 8; i < 32 * HID; i += 32 * 8) *(v8h*)(hbuf + i) = z;
    }

    const int wb     = wgw0 + wave * 32;   // this wave's local word base
    const int dircol = dir ? HID : 0;

    int lenA[8], lenB[8];
#pragma unroll
    for (int v = 0; v < 8; ++v) {
        lenA[v] = s_len[wave * 32 + v + 8 * hi];
        lenB[v] = s_len[wave * 32 + 16 + v + 8 * hi];
    }

    v16h hf0[4], hf1[4];
#pragma unroll
    for (int c = 0; c < 4; ++c)
#pragma unroll
        for (int i = 0; i < 16; ++i) { hf0[c][i] = (_Float16)0.f; hf1[c][i] = (_Float16)0.f; }

    for (int t = 0; t < T_MAX; ++t) {
        // per-row char ids for this step (time remap for backward dir)
        int ch0[8], ch1[8];
#pragma unroll
        for (int v = 0; v < 8; ++v) {
            int l0 = lenA[v];
            int ts0 = dir ? ((t < l0) ? (l0 - 1 - t) : t) : t;
            ch0[v] = s_ch[(wave * 32 + v + 8 * hi) * T_MAX + ts0];
            int l1 = lenB[v];
            int ts1 = dir ? ((t < l1) ? (l1 - 1 - t) : t) : t;
            ch1[v] = s_ch[(wave * 32 + 16 + v + 8 * hi) * T_MAX + ts1];
        }

#pragma unroll 1
        for (int j = 0; j < 8; ++j) {
            const int col = j * 16 + nl;
            const float br = s_bhh[col];
            const float bz = s_bhh[128 + col];
            const float bn = s_bhh[256 + col];

            v8f ar0, az0, anh0, ar1, az1, anh1;
            float anx0[8], anx1[8];
#pragma unroll
            for (int v = 0; v < 8; ++v) {
                const _Float16* e0 = s_embP + ch0[v] * 384;
                const _Float16* e1 = s_embP + ch1[v] * 384;
                ar0[v]  = br + (float)e0[col];
                az0[v]  = bz + (float)e0[128 + col];
                anx0[v] =      (float)e0[256 + col];
                ar1[v]  = br + (float)e1[col];
                az1[v]  = bz + (float)e1[128 + col];
                anx1[v] =      (float)e1[256 + col];
                anh0[v] = bn;
                anh1[v] = bn;
            }
            // h @ Whh^T: each B-fragment feeds both word tiles
#pragma unroll
            for (int c = 0; c < 4; ++c) {
                int kb = c * 32;
                v16h b = load_frag(s_Whh16, col, HID, kb, hi);
                ar0 = wmma16(hf0[c], b, ar0);
                ar1 = wmma16(hf1[c], b, ar1);
                b = load_frag(s_Whh16, 128 + col, HID, kb, hi);
                az0 = wmma16(hf0[c], b, az0);
                az1 = wmma16(hf1[c], b, az1);
                b = load_frag(s_Whh16, 256 + col, HID, kb, hi);
                anh0 = wmma16(hf0[c], b, anh0);
                anh1 = wmma16(hf1[c], b, anh1);
            }
            // elementwise gates in C layout: lane holds (M = v + 8*hi, N = nl)
#pragma unroll
            for (int v = 0; v < 8; ++v) {
                const int M = v + 8 * hi;
                {   // tile 0
                    float hold = (float)hbuf[M * HID + col];
                    float r = sigm(ar0[v]);
                    float z = sigm(az0[v]);
                    float n = tanhf(anx0[v] + r * anh0[v]);
                    float hn = (1.f - z) * n + z * hold;
                    hbuf[M * HID + col] = (_Float16)hn;
                    int L = lenA[v];
                    int valid = (t < L);
                    int tpos = dir ? (valid ? (L - 1 - t) : t) : t;
                    ws[((size_t)(wb + M) * T_MAX + tpos) * 256 + dircol + col] =
                        (_Float16)(valid ? hn : 0.f);
                }
                {   // tile 1
                    float hold = (float)hbuf[(16 + M) * HID + col];
                    float r = sigm(ar1[v]);
                    float z = sigm(az1[v]);
                    float n = tanhf(anx1[v] + r * anh1[v]);
                    float hn = (1.f - z) * n + z * hold;
                    hbuf[(16 + M) * HID + col] = (_Float16)hn;
                    int L = lenB[v];
                    int valid = (t < L);
                    int tpos = dir ? (valid ? (L - 1 - t) : t) : t;
                    ws[((size_t)(wb + 16 + M) * T_MAX + tpos) * 256 + dircol + col] =
                        (_Float16)(valid ? hn : 0.f);
                }
            }
        }
        // re-form h A-fragments for the next step (same-wave LDS, in-order)
        if (t + 1 < T_MAX) {
#pragma unroll
            for (int c = 0; c < 4; ++c) {
                hf0[c] = load_frag(hbuf,            nl, HID, c * 32, hi);
                hf1[c] = load_frag(hbuf + 16 * HID, nl, HID, c * 32, hi);
            }
        }
    }
}

// ---------------------------------------------------------------------------
// Attention kernel. 8 waves/WG, 16 words per wave. Each per-t tile
// out[16 words, 256] f16 is DMA'd into LDS by the Tensor Data Mover
// (double-buffered, TDM row padding de-conflicts LDS banks), then used for
// both the WMMA projection and the online softmax-weighted accumulation, so
// ws is read from HBM exactly once.
// ---------------------------------------------------------------------------
__device__ inline void stage_issue(const _Float16* __restrict__ ws, int lw0, int t,
                                   unsigned ldsOff, _Float16* stagePtr, int lane)
{
#if HAVE_TDM
    (void)stagePtr; (void)lane;
    unsigned long long ga = (unsigned long long)(size_t)ws +
                            ((unsigned long long)lw0 * T_MAX + (unsigned)t) * 512ull;
    uint32x4 g0;
    g0.x = 1u;                               // count=1, no gather, user descriptor
    g0.y = ldsOff;                           // LDS byte address of tile start
    g0.z = (unsigned)ga;                     // global_addr[31:0]
    g0.w = (unsigned)((ga >> 32) & 0x1FFFFFFu) | (2u << 30);   // addr[56:32] | type=2
    int32x8 g1;
    g1[0] = (int)((1u << 16) |               // data_size = 2B
                  (1u << 20) |               // pad_enable
                  (6u << 22) |               // pad_interval: 128 DWORDs (512B rows)
                  (3u << 25));               // pad_amount: 4 DWORDs (16B)
    g1[1] = (int)(256u << 16);               // tensor_dim0 = 256 (low 16 bits)
    g1[2] = (int)(16u << 16);                // tensor_dim0 hi=0 | tensor_dim1 = 16
    g1[3] = (int)(256u << 16);               // tensor_dim1 hi=0 | tile_dim0 = 256
    g1[4] = 16;                              // tile_dim1 = 16, tile_dim2 = 0
    g1[5] = T_MAX * 256;                     // tensor_dim0_stride = 5120 elements
    g1[6] = 0;
    g1[7] = 0;
    int32x4 z4 = {0, 0, 0, 0};
#if __clang_major__ >= 23
    int32x8 z8 = {0, 0, 0, 0, 0, 0, 0, 0};
    __builtin_amdgcn_tensor_load_to_lds(g0, g1, z4, z4, z8, 0);
#else
    __builtin_amdgcn_tensor_load_to_lds(g0, g1, z4, z4, 0);
#endif
#else
    (void)ldsOff;
    // cooperative synchronous copy with the same 264-half row pitch
    for (int s = lane; s < 16 * 32; s += 32) {
        int r = s >> 5, sg = s & 31;
        const v8h* src = (const v8h*)(ws + ((size_t)(lw0 + r) * T_MAX + t) * 256 + sg * 8);
        *(v8h*)(stagePtr + r * SPITCH + sg * 8) = *src;
    }
#endif
}

__global__ __launch_bounds__(256)
void attn_kernel(const _Float16* __restrict__ ws, const float* __restrict__ Wp,
                 const float* __restrict__ bp, const float* __restrict__ ctx,
                 float* __restrict__ out, int word_base)
{
    extern __shared__ char smem[];
    _Float16* s_Wp  = (_Float16*)(smem);            // 128*256 f16 = 65536 B
    float*    s_bp  = (float*)(smem + 65536);       // 128 f32
    float*    s_ctx = (float*)(smem + 66048);       // 128 f32
    float*    s_sct = (float*)(smem + 66560);       // 8 waves * 16 f32
    const unsigned STAGE_OFF = 67072;               // 8 waves * 2 bufs * 8448 B

    const int tid = threadIdx.x;
    for (int i = tid; i < 128 * 256; i += 256) s_Wp[i] = (_Float16)Wp[i];
    if (tid < 128) { s_bp[tid] = bp[tid]; s_ctx[tid] = ctx[tid]; }
    __syncthreads();

    const int wave = tid >> 5;
    const int lane = tid & 31;
    const int nl   = lane & 15;
    const int hi   = lane >> 4;
    const int lw0  = blockIdx.x * 128 + wave * 16;

    const unsigned myStage0 = STAGE_OFF + (unsigned)wave * 2u * 8448u;
    _Float16* stage[2];
    stage[0] = (_Float16*)(smem + myStage0);
    stage[1] = (_Float16*)(smem + myStage0 + 8448u);

    // prime the double buffer
    stage_issue(ws, lw0, 0, myStage0,         stage[0], lane);
    stage_issue(ws, lw0, 1, myStage0 + 8448u, stage[1], lane);

    float accd[128];    // this lane: word m = nl, dims d = hi*128 .. hi*128+127
#pragma unroll
    for (int k = 0; k < 128; ++k) accd[k] = 0.f;
    float ssum = 0.f;

    for (int t = 0; t < T_MAX; ++t) {
        _Float16* buf = stage[t & 1];
#if HAVE_TDM
        if (t < T_MAX - 2) __builtin_amdgcn_s_wait_tensorcnt(1);
        else               __builtin_amdgcn_s_wait_tensorcnt(0);
#endif
        // A fragments from the staged tile (row pitch SPITCH halves)
        v16h af[8];
#pragma unroll
        for (int c = 0; c < 8; ++c) af[c] = load_frag(buf, nl, SPITCH, c * 32, hi);

        float part[8];
#pragma unroll
        for (int v = 0; v < 8; ++v) part[v] = 0.f;

#pragma unroll 1
        for (int j = 0; j < 8; ++j) {
            v8f acc = splat8(s_bp[j * 16 + nl]);
#pragma unroll
            for (int c = 0; c < 8; ++c)
                acc = wmma16(af[c], load_frag(s_Wp, j * 16 + nl, 256, c * 32, hi), acc);
            float cv = s_ctx[j * 16 + nl];
#pragma unroll
            for (int v = 0; v < 8; ++v) part[v] += tanhf(acc[v]) * cv;
        }
        // reduce score partials across the 16 lanes of each half-wave
#pragma unroll
        for (int v = 0; v < 8; ++v) {
            float s = part[v];
            for (int m = 1; m < 16; m <<= 1) s += __shfl_xor(s, m, 32);
            part[v] = s;
        }
        if (nl == 0) {
#pragma unroll
            for (int v = 0; v < 8; ++v) s_sct[wave * 16 + v + 8 * hi] = part[v];
        }
        float sc = s_sct[wave * 16 + nl];   // score of this lane's word at time t

        // online (shift-free) softmax accumulation; |score| <= sum|ctx| ~ 12.8
        float p = __expf(sc);
        ssum += p;
        {
            const _Float16* rowp = buf + nl * SPITCH + hi * 128;
#pragma unroll
            for (int k = 0; k < 16; ++k) {
                v8h w = *(const v8h*)(rowp + k * 8);
#pragma unroll
                for (int i = 0; i < 8; ++i) accd[k * 8 + i] += p * (float)w[i];
            }
        }
        // refill the buffer we just consumed
        if (t + 2 < T_MAX) {
            asm volatile("s_wait_dscnt 0x0" ::: "memory");
            stage_issue(ws, lw0, t + 2, myStage0 + (unsigned)(t & 1) * 8448u,
                        stage[t & 1], lane);
        }
    }

    const float inv = 1.f / ssum;
    float* o = out + (size_t)(word_base + lw0 + nl) * 256 + hi * 128;
#pragma unroll
    for (int k = 0; k < 128; ++k) o[k] = accd[k] * inv;
}

// ---------------------------------------------------------------------------
extern "C" void kernel_launch(void* const* d_in, const int* in_sizes, int n_in,
                              void* d_out, int out_size, void* d_ws, size_t ws_size,
                              hipStream_t stream)
{
    (void)in_sizes; (void)n_in; (void)out_size;
    const int*   chars = (const int*)d_in[0];
    const int*   lens  = (const int*)d_in[1];
    const float* emb   = (const float*)d_in[2];
    const float* Wih_f = (const float*)d_in[3];
    const float* Whh_f = (const float*)d_in[4];
    const float* bih_f = (const float*)d_in[5];
    const float* bhh_f = (const float*)d_in[6];
    const float* Wih_b = (const float*)d_in[7];
    const float* Whh_b = (const float*)d_in[8];
    const float* bih_b = (const float*)d_in[9];
    const float* bhh_b = (const float*)d_in[10];
    const float* Wp    = (const float*)d_in[11];
    const float* bp    = (const float*)d_in[12];
    const float* ctx   = (const float*)d_in[13];
    float*     out = (float*)d_out;
    _Float16*  ws  = (_Float16*)d_ws;

    // chunk word count so the [chunk][T][256] f16 seq-output fits in d_ws
    const size_t perWord = (size_t)T_MAX * 256 * sizeof(_Float16); // 10240 B
    long long chunk = (long long)(ws_size / perWord);
    if (chunk > B_TOT) chunk = B_TOT;
    chunk &= ~255LL;                 // GRU WG covers 256 words
    if (chunk < 256) chunk = 256;

    const size_t gruSmem  = 285184;
    const size_t attnSmem = 67072 + 8 * 2 * 8448;   // 202240

    for (int base = 0; base < B_TOT; base += (int)chunk) {
        int nb = B_TOT - base; if (nb > (int)chunk) nb = (int)chunk;
        int wgsG = nb / 256;
        int wgsA = nb / 128;
        hipLaunchKernelGGL(gru_dir_kernel, dim3(wgsG), dim3(256), gruSmem, stream,
                           chars, lens, emb, Wih_f, Whh_f, bih_f, bhh_f, ws, base, 0);
        hipLaunchKernelGGL(gru_dir_kernel, dim3(wgsG), dim3(256), gruSmem, stream,
                           chars, lens, emb, Wih_b, Whh_b, bih_b, bhh_b, ws, base, 1);
        hipLaunchKernelGGL(attn_kernel, dim3(wgsA), dim3(256), attnSmem, stream,
                           ws, Wp, bp, ctx, out, base);
    }
}
